// CTRNN_61366492725335
// MI455X (gfx1250) — matrix operand order repaired and checked by reading, not compiled, not run on hardware
//
#include <hip/hip_runtime.h>

// ---------------------------------------------------------------------------
// CDNA5 (gfx1250) Elman RNN:  h_t = tanh(x_t @ Win^T + h_{t-1} @ Whid^T + b)
//  Phase 0: one-time f32->bf16 conversion of Win/Whid into workspace
//  Phase 1: big WMMA GEMM  Z = X @ Win^T + b   (time-parallel, HBM-bound)
//  Phase 2: 512 sequential WMMA step kernels; Whid-bf16 (8MB) L2-resident,
//           h carried as bf16 ping-pong so the hot loop has zero converts.
//  K-loops are explicitly double-buffered (raw loads for k+1 issued before
//  consuming fragments of k) so s_wait_loadcnt does not serialize each WMMA.
// ---------------------------------------------------------------------------

typedef __attribute__((ext_vector_type(16))) __bf16 v16bf;
typedef __attribute__((ext_vector_type(8)))  __bf16 bf16x8;
typedef __attribute__((ext_vector_type(8)))  float  v8f;
typedef __attribute__((ext_vector_type(4)))  float  f32x4;

// ---- raw (unconverted) fragment payloads ----------------------------------
struct RawF32x16 { f32x4 x0, x1, x2, x3; };   // 16 f32 (A or B frag source)
struct RawBf16A  { bf16x8 x0, x1; };          // 2x b128
struct RawBf16B  { v16bf  x; };               // 2x b128 contiguous

// ---- A-fragment (16x32 bf16, M x K) ---------------------------------------
//   lanes 0-15 : row m0+l, elems 0..15 = K k0..k0+15
//   lanes16-31 : row m0+l, elems 0..15 = K k0+16..k0+31
__device__ __forceinline__ RawF32x16 load_a_raw(const float* __restrict__ A,
                                                int ld, int m0, int k0) {
    const int lane = threadIdx.x & 31, l = lane & 15, hi = lane >> 4;
    const float* r = A + (size_t)(m0 + l) * ld;
    RawF32x16 o;
    o.x0 = ((const f32x4*)(r + k0 + hi * 16))[0];
    o.x1 = ((const f32x4*)(r + k0 + hi * 16))[1];
    o.x2 = ((const f32x4*)(r + k0 + hi * 16))[2];
    o.x3 = ((const f32x4*)(r + k0 + hi * 16))[3];
    return o;
}

__device__ __forceinline__ RawBf16A load_a_raw(const __bf16* __restrict__ A,
                                               int ld, int m0, int k0) {
    const int lane = threadIdx.x & 31, l = lane & 15, hi = lane >> 4;
    const __bf16* r = A + (size_t)(m0 + l) * ld;
    RawBf16A o;
    o.x0 = *(const bf16x8*)(r + k0 + hi * 16);
    o.x1 = *(const bf16x8*)(r + k0 + hi * 16 + 8);
    return o;
}

// ---- B-fragment (32x16 bf16, K x N) for Y = A @ W^T: elem (k,n) = W[n,k] --
__device__ __forceinline__ RawF32x16 load_b_raw(const float* __restrict__ W,
                                                int ld, int n0, int k0) {
    const int lane = threadIdx.x & 31, l = lane & 15, hi = lane >> 4;
    const f32x4* p = (const f32x4*)(W + (size_t)(n0 + l) * ld + k0 + hi * 16);
    RawF32x16 o;
    o.x0 = p[0]; o.x1 = p[1]; o.x2 = p[2]; o.x3 = p[3];
    return o;
}

__device__ __forceinline__ RawBf16B load_b_raw(const __bf16* __restrict__ W,
                                               int ld, int n0, int k0) {
    const int lane = threadIdx.x & 31, l = lane & 15, hi = lane >> 4;
    RawBf16B o;
    o.x = *(const v16bf*)(W + (size_t)(n0 + l) * ld + k0 + hi * 16);
    return o;
}

// ---- convert raw payload -> WMMA operand at consume time ------------------
__device__ __forceinline__ v16bf make_frag(const RawF32x16& r) {
    v16bf a;
#pragma unroll
    for (int i = 0; i < 4; ++i) {
        a[i]      = (__bf16)r.x0[i];
        a[4 + i]  = (__bf16)r.x1[i];
        a[8 + i]  = (__bf16)r.x2[i];
        a[12 + i] = (__bf16)r.x3[i];
    }
    return a;
}
__device__ __forceinline__ v16bf make_frag(const RawBf16A& r) {
    return __builtin_shufflevector(r.x0, r.x1, 0, 1, 2, 3, 4, 5, 6, 7,
                                   8, 9, 10, 11, 12, 13, 14, 15);
}
__device__ __forceinline__ v16bf make_frag(const RawBf16B& r) { return r.x; }

#define WMMA_BF16(a, b, c) \
    __builtin_amdgcn_wmma_f32_16x16x32_bf16(false, (a), false, (b), (short)0, (c), false, false)

// ---------------------------------------------------------------------------
// Phase 0: one-time f32 -> bf16 (8 elements per thread, b128 in / b128 out)
// ---------------------------------------------------------------------------
__global__ __launch_bounds__(256) void cvt_f32_to_bf16(
    const float* __restrict__ s, __bf16* __restrict__ d, int n8) {
    int i = blockIdx.x * blockDim.x + threadIdx.x;
    if (i >= n8) return;
    const f32x4* p = (const f32x4*)s + 2 * (size_t)i;
    f32x4 x0 = p[0], x1 = p[1];
    bf16x8 o;
#pragma unroll
    for (int j = 0; j < 4; ++j) { o[j] = (__bf16)x0[j]; o[4 + j] = (__bf16)x1[j]; }
    ((bf16x8*)d)[i] = o;
}

// ---------------------------------------------------------------------------
// Phase 1: Z[M,H] = X[M,K] @ Wi[H,K]^T + bias[H]   (M = T*B)
// Wave: 16 rows x 64 cols; WG: 64x128. Double-buffered K-loop.
// ---------------------------------------------------------------------------
template <typename BT>
__global__ __launch_bounds__(256) void rnn_input_gemm(
    const float* __restrict__ X, const BT* __restrict__ Wi,
    const float* __restrict__ bias, float* __restrict__ Z, int K, int H) {
    const int wave = threadIdx.x >> 5;
    const int lane = threadIdx.x & 31;
    const int m0 = blockIdx.y * 64 + (wave & 3) * 16;
    const int n0 = blockIdx.x * 128 + (wave >> 2) * 64;

    v8f c[4] = {};
    auto ar  = load_a_raw(X, K, m0, 0);
    auto br0 = load_b_raw(Wi, K, n0,      0);
    auto br1 = load_b_raw(Wi, K, n0 + 16, 0);
    auto br2 = load_b_raw(Wi, K, n0 + 32, 0);
    auto br3 = load_b_raw(Wi, K, n0 + 48, 0);

    for (int k0 = 0; k0 < K; k0 += 32) {
        const int kn = (k0 + 32 < K) ? (k0 + 32) : 0;   // branchless tail
        auto arn  = load_a_raw(X, K, m0, kn);
        auto brn0 = load_b_raw(Wi, K, n0,      kn);
        auto brn1 = load_b_raw(Wi, K, n0 + 16, kn);
        auto brn2 = load_b_raw(Wi, K, n0 + 32, kn);
        auto brn3 = load_b_raw(Wi, K, n0 + 48, kn);

        v16bf a = make_frag(ar);
        c[0] = WMMA_BF16(a, make_frag(br0), c[0]);
        c[1] = WMMA_BF16(a, make_frag(br1), c[1]);
        c[2] = WMMA_BF16(a, make_frag(br2), c[2]);
        c[3] = WMMA_BF16(a, make_frag(br3), c[3]);

        ar = arn; br0 = brn0; br1 = brn1; br2 = brn2; br3 = brn3;
    }
    // C/D f32 layout: VGPR r -> row m0 + (lane>=16 ? 8 : 0) + r, col n0 + lane%16
    const int l = lane & 15, hi = lane >> 4, row = m0 + hi * 8;
#pragma unroll
    for (int j = 0; j < 4; ++j) {
        const int col = n0 + 16 * j + l;
        const float bv = bias[col];
#pragma unroll
        for (int r = 0; r < 8; ++r)
            Z[(size_t)(row + r) * H + col] = c[j][r] + bv;
    }
}

// h_0 = tanh(Z_0); also emit bf16 copy for the first recurrent step.
__global__ void rnn_h0(float* __restrict__ Z0, __bf16* __restrict__ h0bf, int n) {
    int i = blockIdx.x * blockDim.x + threadIdx.x;
    if (i >= n) return;
    float v = tanhf(Z0[i]);
    Z0[i] = v;
    if (h0bf) h0bf[i] = (__bf16)v;
}

// ---------------------------------------------------------------------------
// Phase 2 (one launch per t): ZH[64,H] <- tanh(ZH + hprev @ Wh^T)
// Wave: 16x32 -> 32 WGs/step. Double-buffered K-loop (critical: these 512
// launches are the serial latency chain). Writes h_t f32 + bf16.
// ---------------------------------------------------------------------------
template <typename AT, typename BT>
__global__ __launch_bounds__(256) void rnn_step(
    const AT* __restrict__ hprev, const BT* __restrict__ Wh,
    float* __restrict__ ZH, __bf16* __restrict__ hout_bf, int H) {
    const int wave = threadIdx.x >> 5;
    const int lane = threadIdx.x & 31;
    const int m0 = (wave & 3) * 16;                     // B = 64 -> 4 M-tiles
    const int n0 = blockIdx.x * 64 + (wave >> 2) * 32;

    v8f c[2] = {};
    auto ar  = load_a_raw(hprev, H, m0, 0);
    auto br0 = load_b_raw(Wh, H, n0,      0);
    auto br1 = load_b_raw(Wh, H, n0 + 16, 0);

    for (int k0 = 0; k0 < H; k0 += 32) {
        const int kn = (k0 + 32 < H) ? (k0 + 32) : 0;
        auto arn  = load_a_raw(hprev, H, m0, kn);
        auto brn0 = load_b_raw(Wh, H, n0,      kn);
        auto brn1 = load_b_raw(Wh, H, n0 + 16, kn);

        v16bf a = make_frag(ar);
        c[0] = WMMA_BF16(a, make_frag(br0), c[0]);
        c[1] = WMMA_BF16(a, make_frag(br1), c[1]);

        ar = arn; br0 = brn0; br1 = brn1;
    }
    const int l = lane & 15, hi = lane >> 4, row = m0 + hi * 8;
#pragma unroll
    for (int j = 0; j < 2; ++j) {
        const int col = n0 + 16 * j + l;
#pragma unroll
        for (int r = 0; r < 8; ++r) {
            const size_t idx = (size_t)(row + r) * H + col;
            float h = tanhf(ZH[idx] + c[j][r]);
            ZH[idx] = h;
            if (hout_bf) hout_bf[idx] = (__bf16)h;
        }
    }
}

// ---------------------------------------------------------------------------
extern "C" void kernel_launch(void* const* d_in, const int* in_sizes, int n_in,
                              void* d_out, int out_size, void* d_ws, size_t ws_size,
                              hipStream_t stream) {
    const float* X  = (const float*)d_in[0];   // [T, B, I]
    const float* Wi = (const float*)d_in[1];   // [H, I]
    const float* Wh = (const float*)d_in[2];   // [H, H]
    const float* bv = (const float*)d_in[3];   // [H]
    float* out = (float*)d_out;                // [T,1,B,H] ++ [1,B,H]

    const int T = 512, B = 64, I = 2048, H = 2048;
    const int M = T * B;
    const size_t step = (size_t)B * H;
    const size_t nWi = (size_t)H * I, nWh = (size_t)H * H;
    const size_t ws_need = 2 * (nWi + nWh + 2 * step);   // ~16.5 MB

    dim3 g1(H / 128, M / 64);                            // input GEMM grid
    const dim3 gs(H / 64);                               // 32 WGs per step

    if (ws_size >= ws_need) {
        // ---- bf16 fast path ------------------------------------------------
        __bf16* Wib = (__bf16*)d_ws;
        __bf16* Whb = Wib + nWi;
        __bf16* hb0 = Whb + nWh;                         // ping-pong h (bf16)
        __bf16* hb1 = hb0 + step;

        cvt_f32_to_bf16<<<(int)(nWi / 8 + 255) / 256, 256, 0, stream>>>(Wi, Wib, (int)(nWi / 8));
        cvt_f32_to_bf16<<<(int)(nWh / 8 + 255) / 256, 256, 0, stream>>>(Wh, Whb, (int)(nWh / 8));

        rnn_input_gemm<__bf16><<<g1, 256, 0, stream>>>(X, Wib, bv, out, I, H);
        rnn_h0<<<(int)((step + 255) / 256), 256, 0, stream>>>(out, hb0, (int)step);
        for (int t = 1; t < T; ++t) {
            __bf16* hp = (t & 1) ? hb0 : hb1;            // h_{t-1}
            __bf16* hn = (t & 1) ? hb1 : hb0;            // h_t
            rnn_step<__bf16, __bf16><<<gs, 256, 0, stream>>>(
                hp, Whb, out + (size_t)t * step, hn, H);
        }
    } else {
        // ---- f32-weight fallback (no workspace needed) ---------------------
        rnn_input_gemm<float><<<g1, 256, 0, stream>>>(X, Wi, bv, out, I, H);
        rnn_h0<<<(int)((step + 255) / 256), 256, 0, stream>>>(out, (__bf16*)nullptr, (int)step);
        for (int t = 1; t < T; ++t) {
            rnn_step<float, float><<<gs, 256, 0, stream>>>(
                out + (size_t)(t - 1) * step, Wh, out + (size_t)t * step,
                (__bf16*)nullptr, H);
        }
    }

    // hx = h_{T-1}, appended after the [T,1,B,H] outputs.
    hipMemcpyAsync(out + (size_t)T * step, out + (size_t)(T - 1) * step,
                   step * sizeof(float), hipMemcpyDeviceToDevice, stream);
}
